// Seq2SeqMulti_2284922602262
// MI455X (gfx1250) — compile-verified, matrix operand order checked by reading, not run on hardware
//
#include <hip/hip_runtime.h>
#include <hip/hip_bf16.h>

// ---------------- problem constants ----------------
#define V_    20000
#define E_    256
#define H_    512
#define B_    64
#define TSRC  200
#define TTRG  128
#define G3H   1536      // 3*H
#define KX    288       // E+1 padded up to multiple of 32
#define KR    768       // E + H  (input to r1)
#define VT    5         // V-tiles per wave in vocab kernel (1250 = 250*5)

typedef __attribute__((ext_vector_type(16))) __bf16 v16bf;
typedef __attribute__((ext_vector_type(8)))  float  v8f;

// ---------------- helpers ----------------
__device__ __forceinline__ unsigned short f32_to_bf16(float f) {
  unsigned int u = __float_as_uint(f);
  u += 0x7FFFu + ((u >> 16) & 1u);          // round-to-nearest-even
  return (unsigned short)(u >> 16);
}

__device__ __forceinline__ float sigm(float x) {
  return 1.0f / (1.0f + __expf(-x));
}

// Load one 16x32 bf16 WMMA operand tile (works for A [M][K] and B stored as W[N][K]).
// Row-major source with leading dim ld (in elements). row0/k0 = tile origin.
// Lane L: row = row0 + (L&15); kb = (L>>4)*8 -> elems 0..7 = K kb..kb+7,
// elems 8..15 = K 16+kb..16+kb+7 (CDNA5 16-bit A/B VGPR layout).
__device__ __forceinline__ v16bf load_tile16(const unsigned short* __restrict__ base,
                                             int ld, int row0, int k0) {
  int lane = threadIdx.x & 31;
  int r  = row0 + (lane & 15);
  int kb = (lane >> 4) * 8;
  const unsigned short* p = base + (size_t)r * ld + k0 + kb;
  union { uint4 q[2]; v16bf v; } u;
  u.q[0] = *(const uint4*)(p);
  u.q[1] = *(const uint4*)(p + 16);
  return u.v;
}

__device__ __forceinline__ v8f wmma_bf16(v16bf a, v16bf b, v8f c) {
  return __builtin_amdgcn_wmma_f32_16x16x32_bf16(
      /*neg_a=*/false, a, /*neg_b=*/false, b,
      /*c_mod=*/(short)0, c, /*reuse_a=*/false, /*reuse_b=*/false);
}

// ---------------- conversion / init kernels ----------------
__global__ void k_cvt_bf16(const float* __restrict__ src, unsigned short* __restrict__ dst, int n) {
  int i = blockIdx.x * 256 + threadIdx.x;
  if (i < n) dst[i] = f32_to_bf16(src[i]);
}

__global__ void k_cvt_pad(const float* __restrict__ src, unsigned short* __restrict__ dst,
                          int rows, int scols, int dcols) {
  int i = blockIdx.x * 256 + threadIdx.x;
  int n = rows * dcols;
  if (i >= n) return;
  int r = i / dcols, c = i - r * dcols;
  dst[i] = (c < scols) ? f32_to_bf16(src[r * scols + c]) : (unsigned short)0;
}

__global__ void k_zero_f32(float* p, int n) {
  int i = blockIdx.x * 256 + threadIdx.x;
  if (i < n) p[i] = 0.0f;
}

__global__ void k_zero_u16(unsigned short* p, int n) {
  int i = blockIdx.x * 256 + threadIdx.x;
  if (i < n) p[i] = 0;
}

// ---------------- encoder GRU step (fused WMMA + gate math) ----------------
// grid (H/16, B/16), block 32 (one wave). Each wave: one 16x16 (b,hcol) tile,
// 3 gate accumulators over K=512 (gh = h @ Whh.T); gi from D_IN=3 scalar FMAs.
__global__ void k_enc_step(const float* __restrict__ hin,
                           const unsigned short* __restrict__ hin_bf,
                           float* __restrict__ hout,
                           unsigned short* __restrict__ hout_bf,
                           const unsigned short* __restrict__ Whh_bf,   // [1536][512]
                           const float* __restrict__ src,               // [T][B][3]
                           const int*   __restrict__ srclen,            // [B]
                           const float* __restrict__ Wih,               // [1536][3]
                           const float* __restrict__ bih,
                           const float* __restrict__ bhh,
                           int t) {
  int n0 = blockIdx.x * 16, m0 = blockIdx.y * 16, lane = threadIdx.x;
  v8f aR = {}, aZ = {}, aN = {};
  for (int k0 = 0; k0 < H_; k0 += 32) {
    v16bf a = load_tile16(hin_bf, H_, m0, k0);
    aR = wmma_bf16(a, load_tile16(Whh_bf, H_, 0 * H_ + n0, k0), aR);
    aZ = wmma_bf16(a, load_tile16(Whh_bf, H_, 1 * H_ + n0, k0), aZ);
    aN = wmma_bf16(a, load_tile16(Whh_bf, H_, 2 * H_ + n0, k0), aN);
  }
  int nc = n0 + (lane & 15);
  int mb = m0 + 8 * (lane >> 4);
  float bhR = bhh[nc], bhZ = bhh[H_ + nc], bhN = bhh[2 * H_ + nc];
  float biR = bih[nc], biZ = bih[H_ + nc], biN = bih[2 * H_ + nc];
  const float* wR = Wih + (size_t)nc * 3;
  const float* wZ = Wih + (size_t)(H_ + nc) * 3;
  const float* wN = Wih + (size_t)(2 * H_ + nc) * 3;
#pragma unroll
  for (int r = 0; r < 8; ++r) {
    int b = mb + r;
    const float* x = src + ((size_t)t * B_ + b) * 3;
    float giR = biR + x[0] * wR[0] + x[1] * wR[1] + x[2] * wR[2];
    float giZ = biZ + x[0] * wZ[0] + x[1] * wZ[1] + x[2] * wZ[2];
    float giN = biN + x[0] * wN[0] + x[1] * wN[1] + x[2] * wN[2];
    float rg = sigm(giR + aR[r] + bhR);
    float zg = sigm(giZ + aZ[r] + bhZ);
    float ng = tanhf(giN + rg * (aN[r] + bhN));
    float hold = hin[(size_t)b * H_ + nc];
    float hv = (t < srclen[b]) ? ((1.0f - zg) * ng + zg * hold) : hold;
    hout[(size_t)b * H_ + nc] = hv;
    hout_bf[(size_t)b * H_ + nc] = f32_to_bf16(hv);
  }
}

// ---------------- decoder GRU step ----------------
// gi and gh kept in separate accumulators (n-gate needs inn + r*hn).
__global__ void k_dec_gru(const float* __restrict__ hin,
                          const unsigned short* __restrict__ hin_bf,
                          float* __restrict__ hout,
                          unsigned short* __restrict__ hout_bf,
                          const unsigned short* __restrict__ xpad,     // [B][288] bf16
                          const unsigned short* __restrict__ Wih_bf,   // [1536][288]
                          const unsigned short* __restrict__ Whh_bf,   // [1536][512]
                          const float* __restrict__ bih,
                          const float* __restrict__ bhh) {
  int n0 = blockIdx.x * 16, m0 = blockIdx.y * 16, lane = threadIdx.x;
  v8f iR = {}, iZ = {}, iN = {}, hR = {}, hZ = {}, hN = {};
  for (int k0 = 0; k0 < KX; k0 += 32) {
    v16bf a = load_tile16(xpad, KX, m0, k0);
    iR = wmma_bf16(a, load_tile16(Wih_bf, KX, 0 * H_ + n0, k0), iR);
    iZ = wmma_bf16(a, load_tile16(Wih_bf, KX, 1 * H_ + n0, k0), iZ);
    iN = wmma_bf16(a, load_tile16(Wih_bf, KX, 2 * H_ + n0, k0), iN);
  }
  for (int k0 = 0; k0 < H_; k0 += 32) {
    v16bf a = load_tile16(hin_bf, H_, m0, k0);
    hR = wmma_bf16(a, load_tile16(Whh_bf, H_, 0 * H_ + n0, k0), hR);
    hZ = wmma_bf16(a, load_tile16(Whh_bf, H_, 1 * H_ + n0, k0), hZ);
    hN = wmma_bf16(a, load_tile16(Whh_bf, H_, 2 * H_ + n0, k0), hN);
  }
  int nc = n0 + (lane & 15);
  int mb = m0 + 8 * (lane >> 4);
  float biR = bih[nc], biZ = bih[H_ + nc], biN = bih[2 * H_ + nc];
  float bhR = bhh[nc], bhZ = bhh[H_ + nc], bhN = bhh[2 * H_ + nc];
#pragma unroll
  for (int r = 0; r < 8; ++r) {
    int b = mb + r;
    float rg = sigm(iR[r] + biR + hR[r] + bhR);
    float zg = sigm(iZ[r] + biZ + hZ[r] + bhZ);
    float ng = tanhf(iN[r] + biN + rg * (hN[r] + bhN));
    float hold = hin[(size_t)b * H_ + nc];
    float hv = (1.0f - zg) * ng + zg * hold;
    hout[(size_t)b * H_ + nc] = hv;
    hout_bf[(size_t)b * H_ + nc] = f32_to_bf16(hv);
  }
}

// ---------------- vocab projection: logits = h @ pe_W.T + pe_b ----------------
// grid (V/(16*VT), B/16), one wave per block. Each wave computes VT consecutive
// 16x16 V-tiles, reusing one A (h) tile per K-chunk across all VT B-tiles.
__global__ void k_vocab(const unsigned short* __restrict__ h_bf,
                        const unsigned short* __restrict__ peW_bf,     // [20000][512]
                        const float* __restrict__ peb,
                        float* __restrict__ logits) {                  // [B][V]
  int v0 = blockIdx.x * (16 * VT), m0 = blockIdx.y * 16, lane = threadIdx.x;
  v8f acc[VT];
#pragma unroll
  for (int j = 0; j < VT; ++j) acc[j] = (v8f){};
  for (int k0 = 0; k0 < H_; k0 += 32) {
    v16bf a = load_tile16(h_bf, H_, m0, k0);
#pragma unroll
    for (int j = 0; j < VT; ++j)
      acc[j] = wmma_bf16(a, load_tile16(peW_bf, H_, v0 + 16 * j, k0), acc[j]);
  }
  int mb = m0 + 8 * (lane >> 4);
#pragma unroll
  for (int j = 0; j < VT; ++j) {
    int vc = v0 + 16 * j + (lane & 15);
    float bv = peb[vc];
#pragma unroll
    for (int r = 0; r < 8; ++r)
      logits[(size_t)(mb + r) * V_ + vc] = acc[j][r] + bv;
  }
}

// ---------------- per-row max / argmax / log-sum-exp, plus xr gather ----------------
__global__ void k_row_softmax(const float* __restrict__ logits,
                              float* __restrict__ rmax, float* __restrict__ rlogsum,
                              int* __restrict__ idxbuf,
                              unsigned short* __restrict__ xr,         // [B][768]
                              const unsigned short* __restrict__ emb_bf,
                              const unsigned short* __restrict__ h_bf) {
  int b = blockIdx.x, tid = threadIdx.x;
  __shared__ float sv[256];
  __shared__ int   si[256];
  const float* row = logits + (size_t)b * V_;
  float m = -1e30f; int arg = 0;
  for (int v = tid; v < V_; v += 256) {
    float x = row[v];
    if (x > m) { m = x; arg = v; }
  }
  sv[tid] = m; si[tid] = arg;
  __syncthreads();
  for (int s = 128; s > 0; s >>= 1) {
    if (tid < s) {
      if (sv[tid + s] > sv[tid] ||
          (sv[tid + s] == sv[tid] && si[tid + s] < si[tid])) {
        sv[tid] = sv[tid + s]; si[tid] = si[tid + s];
      }
    }
    __syncthreads();
  }
  float rowm = sv[0];
  int rowarg = si[0];
  __syncthreads();
  float s = 0.0f;
  for (int v = tid; v < V_; v += 256) s += __expf(row[v] - rowm);
  sv[tid] = s;
  __syncthreads();
  for (int q = 128; q > 0; q >>= 1) {
    if (tid < q) sv[tid] += sv[tid + q];
    __syncthreads();
  }
  if (tid == 0) { rmax[b] = rowm; rlogsum[b] = logf(sv[0]); idxbuf[b] = rowarg; }
  // build xr[b] = [emb[argmax] (bf16), h (bf16)]
  const unsigned short* er = emb_bf + (size_t)rowarg * E_;
  for (int c = tid; c < E_; c += 256) xr[(size_t)b * KR + c] = er[c];
  for (int c = tid; c < H_; c += 256) xr[(size_t)b * KR + E_ + c] = h_bf[(size_t)b * H_ + c];
}

// log_p = logit - max - log(sumexp); vectorized float4 HBM stream (1.3 GB total).
__global__ void k_write_logp(const float4* __restrict__ logits4,
                             const float* __restrict__ rmax,
                             const float* __restrict__ rlogsum,
                             float4* __restrict__ out4) {              // [B][V/4]
  int i = blockIdx.x * 256 + threadIdx.x;
  if (i >= B_ * (V_ / 4)) return;
  int b = i / (V_ / 4);
  float m = rmax[b] + rlogsum[b];
  float4 x = logits4[i];
  x.x -= m; x.y -= m; x.z -= m; x.w -= m;
  out4[i] = x;
}

// ---------------- r1 = relu(concat(emb[idx], h) @ r1_W.T + r1_b) ----------------
__global__ void k_rate1(const unsigned short* __restrict__ xr,          // [B][768]
                        const unsigned short* __restrict__ r1W_bf,      // [512][768]
                        const float* __restrict__ r1b,
                        float* __restrict__ r1o) {                      // [B][512]
  int n0 = blockIdx.x * 16, m0 = blockIdx.y * 16, lane = threadIdx.x;
  v8f acc = {};
  for (int k0 = 0; k0 < KR; k0 += 32)
    acc = wmma_bf16(load_tile16(xr, KR, m0, k0),
                    load_tile16(r1W_bf, KR, n0, k0), acc);
  int nc = n0 + (lane & 15);
  int mb = m0 + 8 * (lane >> 4);
  float bb = r1b[nc];
#pragma unroll
  for (int r = 0; r < 8; ++r) {
    float v = acc[r] + bb;
    r1o[(size_t)(mb + r) * H_ + nc] = v > 0.0f ? v : 0.0f;
  }
}

__global__ void k_rate2(const float* __restrict__ r1o,
                        const float* __restrict__ r2W, const float* __restrict__ r2b,
                        float* __restrict__ ratebuf, float* __restrict__ outslot) {
  int b = blockIdx.x, tid = threadIdx.x;
  __shared__ float sh[256];
  float s = 0.0f;
  for (int c = tid; c < H_; c += 256) s += r1o[(size_t)b * H_ + c] * r2W[c];
  sh[tid] = s;
  __syncthreads();
  for (int q = 128; q > 0; q >>= 1) {
    if (tid < q) sh[tid] += sh[tid + q];
    __syncthreads();
  }
  if (tid == 0) {
    float rr = sigm(sh[0] + r2b[0]);
    ratebuf[b] = rr;
    outslot[b] = rr;
  }
}

__global__ void k_init_dec(const int* __restrict__ trg_eid, const float* __restrict__ trg_rate,
                           int* __restrict__ idxbuf, float* __restrict__ ratebuf) {
  int b = threadIdx.x;
  if (b < B_) { idxbuf[b] = trg_eid[b]; ratebuf[b] = trg_rate[b]; }
}

__global__ void k_build_x(unsigned short* __restrict__ xpad,
                          const unsigned short* __restrict__ emb_bf,
                          const int* __restrict__ idxbuf,
                          const float* __restrict__ ratebuf) {
  int b = blockIdx.x;
  int id = idxbuf[b];
  float rt = ratebuf[b];
  for (int c = threadIdx.x; c < KX; c += blockDim.x) {
    unsigned short val;
    if (c < E_)       val = emb_bf[(size_t)id * E_ + c];
    else if (c == E_) val = f32_to_bf16(rt);
    else              val = 0;
    xpad[(size_t)b * KX + c] = val;
  }
}

// ---------------- host-side orchestration ----------------
extern "C" void kernel_launch(void* const* d_in, const int* in_sizes, int n_in,
                              void* d_out, int out_size, void* d_ws, size_t ws_size,
                              hipStream_t stream) {
  // inputs, setup_inputs() order
  const float* src      = (const float*)d_in[0];   // [200][64][3]
  const int*   src_len  = (const int*)  d_in[1];   // [64]
  const int*   trg_eid  = (const int*)  d_in[2];   // [128][64]
  const float* trg_rate = (const float*)d_in[3];   // [128][64]
  const float* emb      = (const float*)d_in[4];   // [20000][256]
  const float* enc_Wih  = (const float*)d_in[5];   // [1536][3]
  const float* enc_Whh  = (const float*)d_in[6];   // [1536][512]
  const float* enc_bih  = (const float*)d_in[7];
  const float* enc_bhh  = (const float*)d_in[8];
  const float* dec_Wih  = (const float*)d_in[9];   // [1536][257]
  const float* dec_Whh  = (const float*)d_in[10];  // [1536][512]
  const float* dec_bih  = (const float*)d_in[11];
  const float* dec_bhh  = (const float*)d_in[12];
  const float* pe_W     = (const float*)d_in[13];  // [20000][512]
  const float* pe_b     = (const float*)d_in[14];
  const float* r1_W     = (const float*)d_in[15];  // [512][768]
  const float* r1_b     = (const float*)d_in[16];
  const float* r2_W     = (const float*)d_in[17];  // [512]
  const float* r2_b     = (const float*)d_in[18];

  float* out = (float*)d_out;
  const size_t EID_ELEMS = (size_t)TTRG * B_ * V_;   // 163,840,000
  float* out_rate = out + EID_ELEMS;                 // [128][64]

  // workspace layout (256B aligned slabs)
  char* w = (char*)d_ws;
  size_t o = 0;
  auto take = [&](size_t bytes) -> void* {
    void* p = w + o;
    o = (o + bytes + 255) & ~(size_t)255;
    return p;
  };
  unsigned short* encWhh_bf = (unsigned short*)take((size_t)G3H * H_ * 2);
  unsigned short* decWhh_bf = (unsigned short*)take((size_t)G3H * H_ * 2);
  unsigned short* decWih_bf = (unsigned short*)take((size_t)G3H * KX * 2);
  unsigned short* peW_bf    = (unsigned short*)take((size_t)V_ * H_ * 2);
  unsigned short* r1W_bf    = (unsigned short*)take((size_t)H_ * KR * 2);
  unsigned short* emb_bf    = (unsigned short*)take((size_t)V_ * E_ * 2);
  float*          h_f32[2];
  unsigned short* h_bf[2];
  h_f32[0] = (float*)take((size_t)B_ * H_ * 4);
  h_f32[1] = (float*)take((size_t)B_ * H_ * 4);
  h_bf[0]  = (unsigned short*)take((size_t)B_ * H_ * 2);
  h_bf[1]  = (unsigned short*)take((size_t)B_ * H_ * 2);
  unsigned short* xpad   = (unsigned short*)take((size_t)B_ * KX * 2);
  unsigned short* xr     = (unsigned short*)take((size_t)B_ * KR * 2);
  float* logits  = (float*)take((size_t)B_ * V_ * 4);
  float* r1o     = (float*)take((size_t)B_ * H_ * 4);
  float* rmax    = (float*)take(B_ * 4);
  float* rlogsum = (float*)take(B_ * 4);
  float* ratebuf = (float*)take(B_ * 4);
  int*   idxbuf  = (int*)take(B_ * 4);

  // ---- weight conversion (fp32 -> bf16), done every call (deterministic) ----
  auto cvt = [&](const float* s, unsigned short* d, int n) {
    k_cvt_bf16<<<(n + 255) / 256, 256, 0, stream>>>(s, d, n);
  };
  cvt(enc_Whh, encWhh_bf, G3H * H_);
  cvt(dec_Whh, decWhh_bf, G3H * H_);
  cvt(pe_W,    peW_bf,    V_ * H_);
  cvt(r1_W,    r1W_bf,    H_ * KR);
  cvt(emb,     emb_bf,    V_ * E_);
  k_cvt_pad<<<(G3H * KX + 255) / 256, 256, 0, stream>>>(dec_Wih, decWih_bf, G3H, E_ + 1, KX);

  // ---- h0 = 0 ----
  k_zero_f32<<<(B_ * H_ + 255) / 256, 256, 0, stream>>>(h_f32[0], B_ * H_);
  k_zero_u16<<<(B_ * H_ + 255) / 256, 256, 0, stream>>>(h_bf[0],  B_ * H_);

  // ---- output slice 0 = zeros (log_p t=0 and rate t=0) ----
  k_zero_f32<<<(B_ * V_ + 255) / 256, 256, 0, stream>>>(out, B_ * V_);
  k_zero_f32<<<1, 64, 0, stream>>>(out_rate, B_);

  dim3 gruGrid(H_ / 16, B_ / 16);   // (32,4)
  int cur = 0;

  // ---- encoder: 200 sequential GRU steps (double-buffered h) ----
  for (int t = 0; t < TSRC; ++t) {
    k_enc_step<<<gruGrid, 32, 0, stream>>>(
        h_f32[cur], h_bf[cur], h_f32[1 - cur], h_bf[1 - cur],
        encWhh_bf, src, src_len, enc_Wih, enc_bih, enc_bhh, t);
    cur ^= 1;
  }

  // ---- decoder: 127 autoregressive steps ----
  k_init_dec<<<1, 64, 0, stream>>>(trg_eid, trg_rate, idxbuf, ratebuf);

  dim3 vocabGrid(V_ / (16 * VT), B_ / 16);  // (250,4)
  for (int s = 0; s < TTRG - 1; ++s) {
    k_build_x<<<B_, 128, 0, stream>>>(xpad, emb_bf, idxbuf, ratebuf);
    k_dec_gru<<<gruGrid, 32, 0, stream>>>(
        h_f32[cur], h_bf[cur], h_f32[1 - cur], h_bf[1 - cur],
        xpad, decWih_bf, decWhh_bf, dec_bih, dec_bhh);
    cur ^= 1;
    k_vocab<<<vocabGrid, 32, 0, stream>>>(h_bf[cur], peW_bf, pe_b, logits);
    k_row_softmax<<<B_, 256, 0, stream>>>(logits, rmax, rlogsum, idxbuf,
                                          xr, emb_bf, h_bf[cur]);
    k_write_logp<<<(B_ * (V_ / 4) + 255) / 256, 256, 0, stream>>>(
        (const float4*)logits, rmax, rlogsum,
        (float4*)(out + (size_t)(s + 1) * B_ * V_));
    k_rate1<<<gruGrid, 32, 0, stream>>>(xr, r1W_bf, r1_b, r1o);
    k_rate2<<<B_, 256, 0, stream>>>(r1o, r2_W, r2_b, ratebuf,
                                    out_rate + (size_t)(s + 1) * B_);
  }

  (void)in_sizes; (void)n_in; (void)out_size; (void)ws_size;
}
